// RowLSTM_52656299049176
// MI455X (gfx1250) — matrix operand order repaired and checked by reading, not compiled, not run on hardware
//
#include <hip/hip_runtime.h>
#include <hip/hip_fp16.h>

typedef __attribute__((ext_vector_type(16))) _Float16 v16h;
typedef __attribute__((ext_vector_type(8)))  float    v8f;

#define B_    8
#define C_    32
#define H_    256
#define W_    256
#define HID_  64
#define HP_   (H_ + 2)          // xt rows incl. zero halo
#define ROWF_ (W_ * C_)         // f16 per (b,h') row of xt = 8192

#define L2E_  1.4426950408889634f

// ---------------------------------------------------------------------------
// Zero the H-halo rows of xt (row 0 and row 257 for each b).
__global__ void halo_zero_kernel(unsigned int* __restrict__ xtu) {
    int tid  = blockIdx.x * blockDim.x + threadIdx.x;   // 0..65535
    int b    = tid >> 13;
    int rsel = (tid >> 12) & 1;
    int off  = tid & 4095;
    int row  = rsel ? (HP_ - 1) : 0;
    xtu[(size_t)(b * HP_ + row) * (ROWF_ / 2) + off] = 0u;
}

// ---------------------------------------------------------------------------
// x[b][c][h][w] f32  ->  xt[b][h+1][w][c] f16   (c contiguous for WMMA B frags)
__global__ void transpose_cvt_kernel(const float* __restrict__ x,
                                     _Float16* __restrict__ xt) {
    __shared__ _Float16 tile[W_ * 34];          // [w][c], pitch 34 (17 banks)
    const int bh = blockIdx.x;                  // 0..2047
    const int b  = bh >> 8;
    const int h  = bh & 255;
    const int t  = threadIdx.x;                 // w = t

    const float* xp = x + ((size_t)(b * C_) * H_ + h) * W_ + t;
#pragma unroll
    for (int c = 0; c < C_; ++c)
        tile[t * 34 + c] = (_Float16)xp[(size_t)c * H_ * W_];
    __syncthreads();

    _Float16* op = xt + ((size_t)(b * HP_) + (h + 1)) * ROWF_;
#pragma unroll
    for (int k = 0; k < 32; ++k) {
        int idx = t + (k << 8);
        int w = idx >> 5, c = idx & 31;
        op[(w << 5) + c] = tile[w * 34 + c];
    }
}

// ---------------------------------------------------------------------------
// Helpers
__device__ __forceinline__ v16h loadB(const uint4* __restrict__ p, int w) {
    union { v16h v; uint4 q[2]; } u;
    u.q[0] = p[w * 4 + 0];
    u.q[1] = p[w * 4 + 2];
    return u.v;
}

// D = A0*B0 + A1*B1 + A2*B2   (C starts as inline 0)
__device__ __forceinline__ v8f gate_mm(v16h A0, v16h A1, v16h A2,
                                       v16h B0, v16h B1, v16h B2) {
    v8f z = {};
    v8f a = __builtin_amdgcn_wmma_f32_16x16x32_f16(false, A0, false, B0,
                                                   (short)0, z, false, false);
    a = __builtin_amdgcn_wmma_f32_16x16x32_f16(false, A1, false, B1,
                                               (short)0, a, false, false);
    a = __builtin_amdgcn_wmma_f32_16x16x32_f16(false, A2, false, B2,
                                               (short)0, a, false, false);
    return a;
}

// Pack one 16x32 f16 A-fragment (ISA 16-bit A layout) for gate output row `oc`.
__device__ __forceinline__ v16h loadA(const float* __restrict__ wrow,
                                      int kh, int half) {
    union { v16h v; _Float16 e[16]; } u;
#pragma unroll
    for (int v = 0; v < 8; ++v) {
        int c0 = ((v >> 2) << 4) + (half << 3) + ((v & 3) << 1);
        u.e[2 * v + 0] = (_Float16)wrow[(c0 + 0) * 3 + kh];
        u.e[2 * v + 1] = (_Float16)wrow[(c0 + 1) * 3 + kh];
    }
    return u.v;
}

// sigmoid(x + b) with nb = -log2(e)*b :  1 / (1 + exp2(-log2e*x + nb))
__device__ __forceinline__ float sigm(float x, float nb) {
    float e = __builtin_amdgcn_exp2f(fmaf(x, -L2E_, nb));
    return __builtin_amdgcn_rcpf(1.0f + e);
}
// tanh(x + b) with nb2 = -2*log2(e)*b
__device__ __forceinline__ float tanh_b(float x, float nb2) {
    float e2 = __builtin_amdgcn_exp2f(fmaf(x, -2.0f * L2E_, nb2));
    return (1.0f - e2) * __builtin_amdgcn_rcpf(1.0f + e2);
}

// ---------------------------------------------------------------------------
// Fused conv(3x1) -> gates -> serial scan over W -> broadcast store.
// Grid: 128 WGs x 128 threads (4 waves). WG: 16 consecutive (b,h) rows, all hid.
// Wave: 16 hid x {i,f,g} gates in the loop; o gate only for the peeled last col.
__global__ void __launch_bounds__(128)
rowlstm_main_kernel(const _Float16* __restrict__ xt,
                    const float* __restrict__ Wc,   // [256][32][3]
                    const float* __restrict__ bc,   // [256]
                    float* __restrict__ out) {      // [8][64][256][256]
    __shared__ float hstage[HID_][16];

    const int tid  = threadIdx.x;
    const int wv   = tid >> 5;         // 0..3
    const int lane = tid & 31;
    const int hidb = wv << 4;          // 16 hid per wave
    const int m    = lane & 15;        // A row / D column index
    const int half = lane >> 4;

    const int nb = blockIdx.x << 4;    // 16 (b,h) rows per WG
    const int b  = nb >> 8;
    const int hb = nb & 255;

    // ---- A fragments for in-loop gates i(0), f(1), g(3) ----
    v16h Ai[3], Af[3], Ag[3];
#pragma unroll
    for (int kh = 0; kh < 3; ++kh) {
        Ai[kh] = loadA(Wc + (0 * 64 + hidb + m) * (C_ * 3), kh, half);
        Af[kh] = loadA(Wc + (1 * 64 + hidb + m) * (C_ * 3), kh, half);
        Ag[kh] = loadA(Wc + (3 * 64 + hidb + m) * (C_ * 3), kh, half);
    }

    // ---- bias -> exp-domain constants (element r <-> oc row r + 8*half) ----
    v8f nbi, nbf, nbg;
#pragma unroll
    for (int r = 0; r < 8; ++r) {
        int ro = hidb + r + (half << 3);
        nbi[r] = -L2E_ * bc[0 * 64 + ro];
        nbf[r] = -L2E_ * bc[1 * 64 + ro];
        nbg[r] = -2.0f * L2E_ * bc[3 * 64 + ro];
    }

    // ---- per-kh B fragment base pointers into xt ----
    const uint4* pB[3];
#pragma unroll
    for (int kh = 0; kh < 3; ++kh) {
        int hp = hb + m + kh;          // = h + kh - 1 + halo(+1), in [0,257]
        pB[kh] = reinterpret_cast<const uint4*>(
                     xt + ((size_t)(b * HP_) + hp) * ROWF_) + half;
    }

    v8f c_state = {};

    // ---- steady-state scan: columns 0 .. W-2 (no output gate) ----
    for (int w = 0; w < W_ - 1; ++w) {
        v16h B0 = loadB(pB[0], w);
        v16h B1 = loadB(pB[1], w);
        v16h B2 = loadB(pB[2], w);
        if (w + 16 < W_) {
            __builtin_prefetch((const void*)&pB[0][(w + 16) * 4], 0, 3);
            __builtin_prefetch((const void*)&pB[1][(w + 16) * 4], 0, 3);
            __builtin_prefetch((const void*)&pB[2][(w + 16) * 4], 0, 3);
        }
        v8f ai = gate_mm(Ai[0], Ai[1], Ai[2], B0, B1, B2);
        v8f af = gate_mm(Af[0], Af[1], Af[2], B0, B1, B2);
        v8f ag = gate_mm(Ag[0], Ag[1], Ag[2], B0, B1, B2);
#pragma unroll
        for (int r = 0; r < 8; ++r) {
            float ig = sigm(ai[r], nbi[r]);
            float fg = sigm(af[r], nbf[r]);
            float gg = tanh_b(ag[r], nbg[r]);
            c_state[r] = fmaf(fg, c_state[r], ig * gg);
        }
    }

    // ---- peeled last column: also needs output gate ----
    v16h Ao[3];
#pragma unroll
    for (int kh = 0; kh < 3; ++kh)
        Ao[kh] = loadA(Wc + (2 * 64 + hidb + m) * (C_ * 3), kh, half);
    v8f nbo;
#pragma unroll
    for (int r = 0; r < 8; ++r)
        nbo[r] = -L2E_ * bc[2 * 64 + hidb + r + (half << 3)];

    {
        const int w = W_ - 1;
        v16h B0 = loadB(pB[0], w);
        v16h B1 = loadB(pB[1], w);
        v16h B2 = loadB(pB[2], w);
        v8f ai = gate_mm(Ai[0], Ai[1], Ai[2], B0, B1, B2);
        v8f af = gate_mm(Af[0], Af[1], Af[2], B0, B1, B2);
        v8f ag = gate_mm(Ag[0], Ag[1], Ag[2], B0, B1, B2);
        v8f ao = gate_mm(Ao[0], Ao[1], Ao[2], B0, B1, B2);
#pragma unroll
        for (int r = 0; r < 8; ++r) {
            float ig = sigm(ai[r], nbi[r]);
            float fg = sigm(af[r], nbf[r]);
            float gg = tanh_b(ag[r], nbg[r]);
            float c  = fmaf(fg, c_state[r], ig * gg);
            float og = sigm(ao[r], nbo[r]);
            hstage[hidb + r + (half << 3)][m] = og * tanh_b(c, 0.0f);
        }
    }
    __syncthreads();

    // ---- broadcast over W with coalesced float4 stores ----
    for (int t = tid; t < HID_ * 16 * (W_ / 4); t += 128) {
        int wq   = t & 63;          // float4 column
        int pair = t >> 6;
        int hl   = pair & 15;
        int hid  = pair >> 4;
        float v  = hstage[hid][hl];
        size_t o = (((size_t)((b * HID_ + hid) * H_) + (hb + hl)) << 8) + (wq << 2);
        *reinterpret_cast<float4*>(out + o) = make_float4(v, v, v, v);
    }
}

// ---------------------------------------------------------------------------
extern "C" void kernel_launch(void* const* d_in, const int* in_sizes, int n_in,
                              void* d_out, int out_size, void* d_ws, size_t ws_size,
                              hipStream_t stream) {
    const float* x  = (const float*)d_in[0];
    const float* Wc = (const float*)d_in[1];
    const float* bc = (const float*)d_in[2];
    float* out      = (float*)d_out;
    _Float16* xt    = (_Float16*)d_ws;   // needs 8*258*8192*2 B ~= 32.3 MiB

    halo_zero_kernel<<<256, 256, 0, stream>>>((unsigned int*)xt);
    transpose_cvt_kernel<<<B_ * H_, 256, 0, stream>>>(x, xt);
    rowlstm_main_kernel<<<(B_ * H_) / 16, 128, 0, stream>>>(xt, Wc, bc, out);
}
